// TTransformer_29515015258473
// MI455X (gfx1250) — matrix operand order repaired and checked by reading, not compiled, not run on hardware
//
#include <hip/hip_runtime.h>

// ---------------------------------------------------------------------------
// MI455X (gfx1250) transformer decoder forward.
// Strategy: all GEMMs in bf16 WMMA (v_wmma_f32_16x16x32_bf16, f32 accum),
// flash attention (one wave32 per 16-query tile, online softmax, P routed
// through LDS for C-layout -> A-layout repack). Logits write (416 MB) is the
// HBM roofline (~18us @ 23.3 TB/s). A-tiles are staged into LDS with the
// CDNA5 async DMA path (global_load_async_to_lds_b128 + s_wait_asynccnt),
// W-tiles go through VGPRs because they are transposed while staging.
// Weights stay L2-resident: 192 MB L2 >> 16 MB of bf16 weights.
// ---------------------------------------------------------------------------

typedef unsigned short u16;
typedef unsigned int   u32;
typedef unsigned long long u64;

typedef __bf16  bf16_t;
typedef bf16_t  v16bf __attribute__((ext_vector_type(16)));
typedef float   v8f   __attribute__((ext_vector_type(8)));

union Frag {
    v16bf v;
    u32   w[8];
    u16   h[16];
};

__device__ __forceinline__ u16 f32_to_bf16(float f) {
    u32 u = __float_as_uint(f);
    u32 r = u + 0x7fffu + ((u >> 16) & 1u);   // round to nearest even
    return (u16)(r >> 16);
}

__device__ __forceinline__ v8f v8f_zero() {
    v8f z;
#pragma unroll
    for (int i = 0; i < 8; i++) z[i] = 0.0f;
    return z;
}

__device__ __forceinline__ v8f wmma_bf16(const Frag& a, const Frag& b, v8f c) {
    return __builtin_amdgcn_wmma_f32_16x16x32_bf16(
        false, a.v, false, b.v, (short)0, c, false, false);
}

// Async DMA one 16B chunk: global -> LDS (ASYNCcnt-tracked).
__device__ __forceinline__ void async_copy_b128(u32 lds_byte_addr, const void* gptr) {
    u64 ga = (u64)(size_t)gptr;
    asm volatile("global_load_async_to_lds_b128 %0, %1, off"
                 :: "v"(lds_byte_addr), "v"(ga) : "memory");
}
__device__ __forceinline__ void async_wait0() {
    asm volatile("s_wait_asynccnt 0x0" ::: "memory");
}

// ---------------------------------------------------------------------------
// f32 -> bf16 bulk convert (weights / activations)
// ---------------------------------------------------------------------------
__global__ void k_cvt(const float* __restrict__ src, u16* __restrict__ dst, size_t n) {
    size_t i = (size_t)blockIdx.x * 256u + threadIdx.x;
    size_t stride = (size_t)gridDim.x * 256u;
    for (; i < n; i += stride) dst[i] = f32_to_bf16(src[i]);
}

// ---------------------------------------------------------------------------
// Token prep: tgt = [CLS, tgt_input_id]; x = 2*emb[tgt] + pe[t]
// Also emits tokens[] (query-pad mask source) and tgt_ output.
// ---------------------------------------------------------------------------
__global__ __launch_bounds__(256) void k_prep_tgt(
    const int* __restrict__ tgt_in, const float* __restrict__ emb,
    const float* __restrict__ pe,
    float* __restrict__ xF, u16* __restrict__ xH,
    int* __restrict__ tokens, float* __restrict__ tgt_out,
    int Tn, int Vn)
{
    int bt = blockIdx.x;
    int b = bt / Tn, t = bt - b * Tn;
    int token = (t == 0) ? Vn : tgt_in[(size_t)b * (Tn - 1) + (t - 1)];
    int d = threadIdx.x;
    float e = emb[(size_t)token * 256 + d];
    float x = 2.0f * e + pe[(size_t)t * 256 + d];
    xF[(size_t)bt * 256 + d] = x;
    xH[(size_t)bt * 256 + d] = f32_to_bf16(x);
    if (d == 0) {
        tokens[bt] = token;
        tgt_out[bt] = (t == 0) ? 0.0f : (float)token;
    }
}

// enc = where(src_mask, src_embedded, 0) -> bf16
__global__ __launch_bounds__(256) void k_prep_enc(
    const float* __restrict__ srcemb, const int* __restrict__ smask,
    u16* __restrict__ encH, int Sn)
{
    int bs = blockIdx.x, d = threadIdx.x;
    int b = bs / Sn, s = bs - b * Sn;
    float v = smask[(size_t)b * Sn + s] ? srcemb[(size_t)bs * 256 + d] : 0.0f;
    encH[(size_t)bs * 256 + d] = f32_to_bf16(v);
}

// ---------------------------------------------------------------------------
// Tiled bf16 WMMA GEMM: C[M,N] = act(A[M,K] @ W[K,N] + bias)
// Block = 128 threads (4 waves), tile 64x64, BK=32.
// A tile staged by async DMA (global_load_async_to_lds_b128); W staged
// transposed via VGPRs so B-fragment K-pairs are contiguous 32-bit LDS loads.
// Requires M % 64 == 0 (all call sites use M=4096).
// ---------------------------------------------------------------------------
#define BM 64
#define BN 64
#define BK 32

__global__ __launch_bounds__(128) void k_gemm_bf16(
    const u16* __restrict__ A, const u16* __restrict__ W,
    const float* __restrict__ bias,
    float* __restrict__ outF, u16* __restrict__ outH,
    int M, int N, int K, int act)
{
    __shared__ __align__(16) u16 As[BM * BK];      // [m][k]
    __shared__ __align__(16) u16 Bst[BN * BK];     // transposed: [n][k]

    int tid  = threadIdx.x;
    int wave = tid >> 5, lane = tid & 31;
    int g = lane >> 4, ln = lane & 15;
    int m0 = blockIdx.y * BM, n0 = blockIdx.x * BN;
    bool fullN = (n0 + BN <= N);

    u32 ldsA = (u32)(size_t)(&As[0]);   // LDS byte offset of As (low 32 bits)

    v8f acc[4];
#pragma unroll
    for (int nt = 0; nt < 4; nt++) acc[nt] = v8f_zero();

    for (int k0 = 0; k0 < K; k0 += BK) {
        // ---- A tile via async DMA: 64 rows x 64B = 256 x 16B chunks
        {
            int chunk = tid;
#pragma unroll
            for (int c = 0; c < 2; c++, chunk += 128) {
                int r  = chunk >> 2;           // row 0..63
                int c8 = chunk & 3;            // 16B chunk within row
                async_copy_b128(ldsA + (u32)(r * BK + c8 * 8) * 2,
                                A + (size_t)(m0 + r) * K + k0 + c8 * 8);
            }
        }
        // ---- W tile staged transposed (VGPR path; transpose swizzle)
        if (fullN) {
            for (int idx = tid; idx < BK * BN / 2; idx += 128) {
                int r = idx >> 5, c2 = idx & 31;
                u32 val = *(const u32*)(W + (size_t)(k0 + r) * N + n0 + c2 * 2);
                Bst[(c2 * 2)     * BK + r] = (u16)val;
                Bst[(c2 * 2 + 1) * BK + r] = (u16)(val >> 16);
            }
        } else {
            for (int idx = tid; idx < BK * BN / 2; idx += 128) {
                int r = idx >> 5, c2 = idx & 31;
                int gn = n0 + c2 * 2;
                u32 val = 0;
                if (gn + 1 < N)
                    val = *(const u32*)(W + (size_t)(k0 + r) * N + gn);
                Bst[(c2 * 2)     * BK + r] = (u16)val;
                Bst[(c2 * 2 + 1) * BK + r] = (u16)(val >> 16);
            }
        }
        // prefetch next K tile into cache hierarchy
        if (k0 + BK < K) {
            __builtin_prefetch(A + (size_t)(m0 + (tid & 63)) * K + k0 + BK, 0, 1);
            __builtin_prefetch(W + (size_t)(k0 + BK + (tid & 31)) * N + n0, 0, 1);
        }
        async_wait0();          // ASYNCcnt -> 0: A tile landed in LDS
        __syncthreads();

        // A fragment (contiguous k = 16*g + s, same order as B)
        Frag a;
        int am = wave * 16 + ln;
#pragma unroll
        for (int i = 0; i < 8; i++) {
            int kk = g * 16 + i * 2;
            a.w[i] = *(const u32*)(As + am * BK + kk);
        }
#pragma unroll
        for (int nt = 0; nt < 4; nt++) {
            Frag bfr;
            int bn = nt * 16 + ln;
#pragma unroll
            for (int i = 0; i < 8; i++) {
                int kk = g * 16 + i * 2;
                bfr.w[i] = *(const u32*)(Bst + bn * BK + kk);
            }
            acc[nt] = wmma_bf16(a, bfr, acc[nt]);
        }
        __syncthreads();
    }

    // epilogue: bias (+ exact GELU), dual f32/bf16 store
#pragma unroll
    for (int nt = 0; nt < 4; nt++) {
        int gn = n0 + nt * 16 + ln;
        if (gn >= N) continue;
        float bv = bias ? bias[gn] : 0.0f;
#pragma unroll
        for (int r = 0; r < 8; r++) {
            int gm = m0 + wave * 16 + r + 8 * g;
            float v = acc[nt][r] + bv;
            if (act == 1) v = 0.5f * v * (1.0f + erff(v * 0.70710678118654752f));
            size_t off = (size_t)gm * N + gn;
            if (outF) outF[off] = v;
            if (outH) outH[off] = f32_to_bf16(v);
        }
    }
}

// ---------------------------------------------------------------------------
// Flash attention: one wave32 per (b, h, 16-query tile). DH=32 => QK^T is a
// single K=32 WMMA per 16x16 score tile. Online softmax; P repacked through
// LDS (C-layout f32 -> A-layout bf16) for the P@V WMMAs.
// ---------------------------------------------------------------------------
__global__ __launch_bounds__(32) void k_attn(
    const u16* __restrict__ Q, const u16* __restrict__ Kb, const u16* __restrict__ Vb,
    u16* __restrict__ O,
    const int* __restrict__ qtok,    // self: token per query (0 => row masked)
    const int* __restrict__ kmask,   // cross: src mask per key
    int Tq, int Tk, int causal)
{
    const float SCALE = 0.17677669529663687f;   // 1/sqrt(32)
    const float NEG   = -3.0e38f;

    int q0 = blockIdx.x * 16;
    int h  = blockIdx.y;
    int b  = blockIdx.z;
    int lane = threadIdx.x;
    int g = lane >> 4, ln = lane & 15;

    __shared__ __align__(16) u16 pl[16 * 32];

    // Q fragment: rows q0..q0+15, K-dim = DH = 32
    Frag qf;
    {
        size_t qrow = ((size_t)b * Tq + q0 + ln) * 256 + h * 32;
#pragma unroll
        for (int i = 0; i < 8; i++) {
            int kk = g * 16 + i * 2;
            qf.w[i] = *(const u32*)(Q + qrow + kk);
        }
    }

    float m_s[8], l_s[8];
    int rowpad[8];
    v8f o0 = v8f_zero(), o1 = v8f_zero();
#pragma unroll
    for (int r = 0; r < 8; r++) {
        m_s[r] = -__builtin_inff();
        l_s[r] = 0.0f;
        int qr = q0 + r + 8 * g;
        rowpad[r] = (qtok && qtok[(size_t)b * Tq + qr] == 0) ? 1 : 0;
    }

    int jend = causal ? (q0 + 16) : Tk;
    for (int j0 = 0; j0 < jend; j0 += 32) {
        // ---- scores: two 16-key WMMAs
        v8f c0 = v8f_zero(), c1 = v8f_zero();
        {
            Frag kf;
            size_t krow = ((size_t)b * Tk + j0 + ln) * 256 + h * 32;
#pragma unroll
            for (int i = 0; i < 8; i++)
                kf.w[i] = *(const u32*)(Kb + krow + g * 16 + 2 * i);
            c0 = wmma_bf16(qf, kf, c0);
        }
        {
            Frag kf;
            size_t krow = ((size_t)b * Tk + j0 + 16 + ln) * 256 + h * 32;
#pragma unroll
            for (int i = 0; i < 8; i++)
                kf.w[i] = *(const u32*)(Kb + krow + g * 16 + 2 * i);
            c1 = wmma_bf16(qf, kf, c1);
        }

        // ---- masking
        int km0 = 1, km1 = 1;
        if (kmask) {
            km0 = kmask[(size_t)b * Tk + j0 + ln];
            km1 = kmask[(size_t)b * Tk + j0 + 16 + ln];
        }
#pragma unroll
        for (int r = 0; r < 8; r++) {
            int qr = q0 + r + 8 * g;
            float s0 = c0[r] * SCALE, s1 = c1[r] * SCALE;
            if (rowpad[r]) { s0 = NEG; s1 = NEG; }
            if (causal) {
                if (j0 + ln > qr)      s0 = NEG;
                if (j0 + 16 + ln > qr) s1 = NEG;
            }
            if (!km0) s0 = NEG;
            if (!km1) s1 = NEG;
            c0[r] = s0; c1[r] = s1;
        }

        // ---- online softmax (row reductions across 16-lane group)
        float alpha[8];
#pragma unroll
        for (int r = 0; r < 8; r++) {
            float mx = fmaxf(c0[r], c1[r]);
#pragma unroll
            for (int off = 1; off < 16; off <<= 1)
                mx = fmaxf(mx, __shfl_xor(mx, off, 32));
            float mn = fmaxf(m_s[r], mx);
            alpha[r] = __expf(m_s[r] - mn);
            m_s[r] = mn;
            float p0 = __expf(c0[r] - mn);
            float p1 = __expf(c1[r] - mn);
            c0[r] = p0; c1[r] = p1;
            float rs = p0 + p1;
#pragma unroll
            for (int off = 1; off < 16; off <<= 1)
                rs += __shfl_xor(rs, off, 32);
            l_s[r] = l_s[r] * alpha[r] + rs;
            o0[r] *= alpha[r];
            o1[r] *= alpha[r];
        }

        // ---- P: C-layout f32 -> LDS bf16 -> A-layout fragment
        __syncthreads();
#pragma unroll
        for (int r = 0; r < 8; r++) {
            int m = r + 8 * g;
            pl[m * 32 + ln]      = f32_to_bf16(c0[r]);
            pl[m * 32 + 16 + ln] = f32_to_bf16(c1[r]);
        }
        __syncthreads();
        Frag pf;
#pragma unroll
        for (int i = 0; i < 8; i++) {
            int kk = g * 16 + i * 2;
            pf.w[i] = *(const u32*)(pl + ln * 32 + kk);
        }

        // ---- V fragments (B-layout: N = dh half, K pairs over 32 keys)
        Frag vf0, vf1;
#pragma unroll
        for (int i = 0; i < 8; i++) {
            int kk = g * 16 + 2 * i;
            size_t vr = ((size_t)b * Tk + j0 + kk) * 256 + h * 32;
            vf0.h[2 * i]     = Vb[vr + ln];
            vf0.h[2 * i + 1] = Vb[vr + 256 + ln];
            vf1.h[2 * i]     = Vb[vr + 16 + ln];
            vf1.h[2 * i + 1] = Vb[vr + 256 + 16 + ln];
        }
        o0 = wmma_bf16(pf, vf0, o0);
        o1 = wmma_bf16(pf, vf1, o1);
    }

    // ---- normalize + store bf16 (feeds the Wo projection GEMM)
#pragma unroll
    for (int r = 0; r < 8; r++) {
        int qr = q0 + r + 8 * g;
        float inv = 1.0f / l_s[r];
        size_t off = ((size_t)b * Tq + qr) * 256 + h * 32;
        O[off + ln]      = f32_to_bf16(o0[r] * inv);
        O[off + 16 + ln] = f32_to_bf16(o1[r] * inv);
    }
}

// ---------------------------------------------------------------------------
// Residual + LayerNorm (D=256): one block per row, in-place capable.
// ---------------------------------------------------------------------------
__global__ __launch_bounds__(256) void k_add_ln(
    const float* __restrict__ X, const float* __restrict__ Yd,
    const float* __restrict__ gamma, const float* __restrict__ beta,
    float* __restrict__ outF, u16* __restrict__ outH)
{
    __shared__ float red[256];
    int row = blockIdx.x, t = threadIdx.x;
    size_t off = (size_t)row * 256 + t;
    float v = X[off] + Yd[off];
    red[t] = v; __syncthreads();
    for (int s = 128; s > 0; s >>= 1) { if (t < s) red[t] += red[t + s]; __syncthreads(); }
    float mean = red[0] * (1.0f / 256.0f);
    __syncthreads();
    float d = v - mean;
    red[t] = d * d; __syncthreads();
    for (int s = 128; s > 0; s >>= 1) { if (t < s) red[t] += red[t + s]; __syncthreads(); }
    float var = red[0] * (1.0f / 256.0f);
    float y = d * rsqrtf(var + 1e-5f) * gamma[t] + beta[t];
    outF[off] = y;
    outH[off] = f32_to_bf16(y);
}

// ---------------------------------------------------------------------------
// Host orchestration
// ---------------------------------------------------------------------------
static inline void* bump(char*& p, size_t bytes) {
    void* r = (void*)p;
    p += (bytes + 255) & ~(size_t)255;
    return r;
}

extern "C" void kernel_launch(void* const* d_in, const int* in_sizes, int n_in,
                              void* d_out, int out_size, void* d_ws, size_t ws_size,
                              hipStream_t stream) {
    const int Bn = 2, Sn = 2048, Tn = 2048, Dn = 256, Hn = 8, DFFn = 2048, Vn = 25426;
    const int BT = Bn * Tn;           // 4096
    const int BS = Bn * Sn;           // 4096

    const float* src_embedded = (const float*)d_in[0];
    const int*   src_mask     = (const int*)  d_in[1];
    const int*   tgt_in       = (const int*)  d_in[2];
    const float* emb          = (const float*)d_in[3];
    const float* pe           = (const float*)d_in[4];
    const float* sa_wq = (const float*)d_in[5];
    const float* sa_wk = (const float*)d_in[6];
    const float* sa_wv = (const float*)d_in[7];
    const float* sa_wo = (const float*)d_in[8];
    const float* sa_bo = (const float*)d_in[9];
    const float* ca_wq = (const float*)d_in[10];
    const float* ca_wk = (const float*)d_in[11];
    const float* ca_wv = (const float*)d_in[12];
    const float* ca_wo = (const float*)d_in[13];
    const float* ca_bo = (const float*)d_in[14];
    const float* ln1_g = (const float*)d_in[15];
    const float* ln1_b = (const float*)d_in[16];
    const float* ln2_g = (const float*)d_in[17];
    const float* ln2_b = (const float*)d_in[18];
    const float* ln3_g = (const float*)d_in[19];
    const float* ln3_b = (const float*)d_in[20];
    const float* fc1_w = (const float*)d_in[21];
    const float* fc1_b = (const float*)d_in[22];
    const float* fc2_w = (const float*)d_in[23];
    const float* fc2_b = (const float*)d_in[24];
    const float* out_w = (const float*)d_in[25];
    const float* out_b = (const float*)d_in[26];

    float* logits  = (float*)d_out;
    float* tgt_out = logits + (size_t)BT * Vn;

    // workspace carve-out
    char* p = (char*)d_ws;
    u16* wq_h   = (u16*)bump(p, (size_t)Dn * Dn * 2);
    u16* wk_h   = (u16*)bump(p, (size_t)Dn * Dn * 2);
    u16* wv_h   = (u16*)bump(p, (size_t)Dn * Dn * 2);
    u16* wo_h   = (u16*)bump(p, (size_t)Dn * Dn * 2);
    u16* cwq_h  = (u16*)bump(p, (size_t)Dn * Dn * 2);
    u16* cwk_h  = (u16*)bump(p, (size_t)Dn * Dn * 2);
    u16* cwv_h  = (u16*)bump(p, (size_t)Dn * Dn * 2);
    u16* cwo_h  = (u16*)bump(p, (size_t)Dn * Dn * 2);
    u16* fc1_h  = (u16*)bump(p, (size_t)Dn * DFFn * 2);
    u16* fc2_h  = (u16*)bump(p, (size_t)DFFn * Dn * 2);
    u16* outw_h = (u16*)bump(p, (size_t)Dn * Vn * 2);
    float* x_f    = (float*)bump(p, (size_t)BT * Dn * 4);
    u16*   x_h    = (u16*)  bump(p, (size_t)BT * Dn * 2);
    u16*   enc_h  = (u16*)  bump(p, (size_t)BS * Dn * 2);
    u16*   q_h    = (u16*)  bump(p, (size_t)BT * Dn * 2);
    u16*   k_h    = (u16*)  bump(p, (size_t)BS * Dn * 2);
    u16*   v_h    = (u16*)  bump(p, (size_t)BS * Dn * 2);
    u16*   attn_h = (u16*)  bump(p, (size_t)BT * Dn * 2);
    float* proj_f = (float*)bump(p, (size_t)BT * Dn * 4);
    u16*   h1_h   = (u16*)  bump(p, (size_t)BT * DFFn * 2);
    int*   tokens = (int*)  bump(p, (size_t)BT * 4);

    auto cvt = [&](const float* s, u16* d, size_t n) {
        int blocks = (int)((n + 255) / 256);
        if (blocks > 4096) blocks = 4096;
        k_cvt<<<blocks, 256, 0, stream>>>(s, d, n);
    };
    auto gemm = [&](const u16* A, const u16* W, const float* bias,
                    float* oF, u16* oH, int M, int N, int K, int act) {
        dim3 grid((N + BN - 1) / BN, (M + BM - 1) / BM);
        k_gemm_bf16<<<grid, 128, 0, stream>>>(A, W, bias, oF, oH, M, N, K, act);
    };

    // 0) weights -> bf16
    cvt(sa_wq, wq_h,  (size_t)Dn * Dn);
    cvt(sa_wk, wk_h,  (size_t)Dn * Dn);
    cvt(sa_wv, wv_h,  (size_t)Dn * Dn);
    cvt(sa_wo, wo_h,  (size_t)Dn * Dn);
    cvt(ca_wq, cwq_h, (size_t)Dn * Dn);
    cvt(ca_wk, cwk_h, (size_t)Dn * Dn);
    cvt(ca_wv, cwv_h, (size_t)Dn * Dn);
    cvt(ca_wo, cwo_h, (size_t)Dn * Dn);
    cvt(fc1_w, fc1_h, (size_t)Dn * DFFn);
    cvt(fc2_w, fc2_h, (size_t)DFFn * Dn);
    cvt(out_w, outw_h, (size_t)Dn * Vn);

    // 1) token prep + encoder mask-and-cast
    k_prep_tgt<<<BT, 256, 0, stream>>>(tgt_in, emb, pe, x_f, x_h, tokens, tgt_out, Tn, Vn);
    k_prep_enc<<<BS, 256, 0, stream>>>(src_embedded, src_mask, enc_h, Sn);

    // 2) self-attention
    gemm(x_h, wq_h, nullptr, nullptr, q_h, BT, Dn, Dn, 0);
    gemm(x_h, wk_h, nullptr, nullptr, k_h, BT, Dn, Dn, 0);
    gemm(x_h, wv_h, nullptr, nullptr, v_h, BT, Dn, Dn, 0);
    {
        dim3 ag(Tn / 16, Hn, Bn);
        k_attn<<<ag, 32, 0, stream>>>(q_h, k_h, v_h, attn_h, tokens, nullptr, Tn, Tn, 1);
    }
    gemm(attn_h, wo_h, sa_bo, proj_f, nullptr, BT, Dn, Dn, 0);
    k_add_ln<<<BT, 256, 0, stream>>>(x_f, proj_f, ln1_g, ln1_b, x_f, x_h);

    // 3) cross-attention
    gemm(x_h,  cwq_h, nullptr, nullptr, q_h, BT, Dn, Dn, 0);
    gemm(enc_h, cwk_h, nullptr, nullptr, k_h, BS, Dn, Dn, 0);
    gemm(enc_h, cwv_h, nullptr, nullptr, v_h, BS, Dn, Dn, 0);
    {
        dim3 ag(Tn / 16, Hn, Bn);
        k_attn<<<ag, 32, 0, stream>>>(q_h, k_h, v_h, attn_h, nullptr, src_mask, Tn, Sn, 0);
    }
    gemm(attn_h, cwo_h, ca_bo, proj_f, nullptr, BT, Dn, Dn, 0);
    k_add_ln<<<BT, 256, 0, stream>>>(x_f, proj_f, ln2_g, ln2_b, x_f, x_h);

    // 4) FFN (exact GELU fused into fc1 epilogue)
    gemm(x_h, fc1_h, fc1_b, nullptr, h1_h, BT, DFFn, Dn, 1);
    gemm(h1_h, fc2_h, fc2_b, proj_f, nullptr, BT, Dn, DFFn, 0);
    k_add_ln<<<BT, 256, 0, stream>>>(x_f, proj_f, ln3_g, ln3_b, x_f, x_h);

    // 5) output projection straight to d_out (the 416 MB HBM-bound store)
    gemm(x_h, outw_h, out_b, logits, nullptr, BT, Vn, Dn, 0);
}
